// PointTransformer_34840774705550
// MI455X (gfx1250) — compile-verified, matrix operand order checked by reading, not run on hardware
//
#include <hip/hip_runtime.h>

typedef __attribute__((ext_vector_type(16))) _Float16 v16h;
typedef __attribute__((ext_vector_type(8)))  _Float16 v8h;
typedef __attribute__((ext_vector_type(8)))  float    v8f;
typedef __attribute__((ext_vector_type(4)))  float    v4f;

#define KNN 16
#define BSZ 16
#define NRCHUNK 8192   // node rows per edge-pipeline chunk (keeps chunk in L2)

// ===================================================== weight pre-packing ===
// Wf layout (f16, zero padded): t = ((kblk*Npad + col)*2 + hs)*16 + e
// where e=0..15 is the B-fragment element order for lane (col%16, hs=lane>>4):
//   v=e>>1, h=e&1, K = 32*kblk + 2*(v&3) + 16*(v>=4) + 8*hs + h   (ISA 7.12.2)
__global__ void pack_w_f16(const float* __restrict__ W, _Float16* __restrict__ Wf,
                           int K, int N, int Npad, int total)
{
  const int t = blockIdx.x * blockDim.x + threadIdx.x;
  if (t >= total) return;
  const int e    = t & 15;
  const int hs   = (t >> 4) & 1;
  const int col  = (t >> 5) % Npad;
  const int kblk = (t >> 5) / Npad;
  const int v = e >> 1, h = e & 1;
  const int k = 32 * kblk + ((v & 3) << 1) + ((v & 4) << 2) + (hs << 3) + h;
  const float val = (k < K && col < N) ? W[(size_t)k * N + col] : 0.f;
  Wf[t] = (_Float16)val;
}

// zero-pad rows of 3 floats to stride 32 (for K=3 GEMM inputs)
__global__ void pad_rows3(const float* __restrict__ in, float* __restrict__ out, int R)
{
  const int t = blockIdx.x * blockDim.x + threadIdx.x;   // R*32
  if (t >= R * 32) return;
  const int c = t & 31, r = t >> 5;
  out[t] = (c < 3) ? in[r * 3 + c] : 0.f;
}

// ============================================================== WMMA GEMM ===
// C[M,Nd] = act(A[M,Ks] @ W + bias). M % 16 == 0, Ks % 32 == 0 (A zero-padded),
// W pre-packed/padded to Npad. One wave32 -> 16x32 of D (2 wmma / K-step).
__global__ __launch_bounds__(128)
void gemm_wmma_fast(const float* __restrict__ A, const _Float16* __restrict__ Wf,
                    const float* __restrict__ bias, float* __restrict__ C,
                    int M, int Ks, int Nd, int Npad, int relu)
{
  const int lane = threadIdx.x & 31;
  const int wave = threadIdx.x >> 5;
  const int m0 = (blockIdx.x * 4 + wave) * 16;
  const int n0 = blockIdx.y * 32;
  if (m0 >= M) return;                         // wave-uniform
  const int lr = lane & 15;
  const int hs = lane >> 4;
  const bool has2 = (n0 + 16) < Npad;          // uniform

  const float* __restrict__ Arow = A + (size_t)(m0 + lr) * Ks + hs * 8;
  const _Float16* __restrict__ Bp0 = Wf + ((size_t)(n0 + lr) << 5) + (hs << 4);
  const _Float16* __restrict__ Bp1 = Bp0 + (16 << 5);
  const size_t bstep = (size_t)Npad << 5;      // halves per kblk

  v8f acc0 = {}, acc1 = {};
  const int nk = Ks >> 5;
  for (int kb = 0; kb < nk; ++kb) {
    const int k0 = kb << 5;
    if (k0 + 32 < Ks) __builtin_prefetch(Arow + k0 + 32, 0, 1);
    // A fragment: two contiguous 8-float runs -> 4x b128 loads, cvt to f16
    const v4f* ap = (const v4f*)(Arow + k0);
    const v4f r0 = ap[0], r1 = ap[1];
    const v4f* aq = (const v4f*)(Arow + k0 + 16);
    const v4f r2 = aq[0], r3 = aq[1];
    v16h af;
    af[0] = (_Float16)r0[0]; af[1] = (_Float16)r0[1];
    af[2] = (_Float16)r0[2]; af[3] = (_Float16)r0[3];
    af[4] = (_Float16)r1[0]; af[5] = (_Float16)r1[1];
    af[6] = (_Float16)r1[2]; af[7] = (_Float16)r1[3];
    af[8]  = (_Float16)r2[0]; af[9]  = (_Float16)r2[1];
    af[10] = (_Float16)r2[2]; af[11] = (_Float16)r2[3];
    af[12] = (_Float16)r3[0]; af[13] = (_Float16)r3[1];
    af[14] = (_Float16)r3[2]; af[15] = (_Float16)r3[3];
    // B fragments: pre-packed, 2x b128 each
    const v8h* bp = (const v8h*)(Bp0 + kb * bstep);
    const v8h b0 = bp[0], b1 = bp[1];
    const v16h bf0 = __builtin_shufflevector(b0, b1, 0,1,2,3,4,5,6,7,8,9,10,11,12,13,14,15);
    acc0 = __builtin_amdgcn_wmma_f32_16x16x32_f16(false, af, false, bf0,
                                                  (short)0, acc0, false, false);
    if (has2) {
      const v8h* bq = (const v8h*)(Bp1 + kb * bstep);
      const v8h c0 = bq[0], c1 = bq[1];
      const v16h bf1 = __builtin_shufflevector(c0, c1, 0,1,2,3,4,5,6,7,8,9,10,11,12,13,14,15);
      acc1 = __builtin_amdgcn_wmma_f32_16x16x32_f16(false, af, false, bf1,
                                                    (short)0, acc1, false, false);
    }
  }

  // store: rows always valid (M%16==0); guard columns vs Nd only
  const int col0 = n0 + lr;
  const int col1 = col0 + 16;
  if (col0 < Nd) {
    const float bv = bias ? bias[col0] : 0.f;
#pragma unroll
    for (int r = 0; r < 8; ++r) {
      const int row = m0 + r + 8 * hs;         // C/D layout: VGPR r -> M=r(+8)
      float v = acc0[r] + bv;
      if (relu) v = fmaxf(v, 0.f);
      C[(size_t)row * Nd + col0] = v;
    }
  }
  if (has2 && col1 < Nd) {
    const float bv = bias ? bias[col1] : 0.f;
#pragma unroll
    for (int r = 0; r < 8; ++r) {
      const int row = m0 + r + 8 * hs;
      float v = acc1[r] + bv;
      if (relu) v = fmaxf(v, 0.f);
      C[(size_t)row * Nd + col1] = v;
    }
  }
}

// ================================================================== kNN =====
__global__ __launch_bounds__(256)
void knn_kernel(const float* __restrict__ qpos, const float* __restrict__ bpos,
                int* __restrict__ nbr, int Mq, int Nb, int exclude_self)
{
  __shared__ float sx[1024], sy[1024], sz[1024];
  const int b  = blockIdx.y;
  const int qi = blockIdx.x * blockDim.x + threadIdx.x;
  const float* qp = qpos + ((size_t)b * Mq + (qi < Mq ? qi : 0)) * 3;
  const float qx = qp[0], qy = qp[1], qz = qp[2];
  float bd[KNN]; int bi[KNN];
#pragma unroll
  for (int t = 0; t < KNN; ++t) { bd[t] = 3.0e38f; bi[t] = 0; }

  for (int j0 = 0; j0 < Nb; j0 += 1024) {
    const int tile = (Nb - j0 < 1024) ? (Nb - j0) : 1024;
    __syncthreads();
    for (int t = threadIdx.x; t < tile; t += blockDim.x) {
      const float* pp = bpos + ((size_t)b * Nb + j0 + t) * 3;
      sx[t] = pp[0]; sy[t] = pp[1]; sz[t] = pp[2];
    }
    __syncthreads();
    for (int t = 0; t < tile; ++t) {
      const int j = j0 + t;
      const float dx = qx - sx[t], dy = qy - sy[t], dz = qz - sz[t];
      float d = dx * dx + dy * dy + dz * dz;
      if (exclude_self && j == qi) d = 3.0e38f;
      if (d < bd[KNN - 1]) {
        int p = KNN - 1;
        while (p > 0 && bd[p - 1] > d) { bd[p] = bd[p - 1]; bi[p] = bi[p - 1]; --p; }
        bd[p] = d; bi[p] = j;
      }
    }
  }
  if (qi < Mq) {
    int* out = nbr + ((size_t)b * Mq + qi) * KNN;
#pragma unroll
    for (int t = 0; t < KNN; ++t) out[t] = bi[t];
  }
}

// ================================================================== FPS =====
__global__ __launch_bounds__(256)
void fps_kernel(const float* __restrict__ pos, int* __restrict__ idx,
                float* __restrict__ dbuf, int N, int m)
{
  __shared__ float sred[256];
  __shared__ int   sidx[256];
  __shared__ float cur[3];
  const int b = blockIdx.x, tid = threadIdx.x;
  const float* P = pos + (size_t)b * N * 3;
  float* d = dbuf + (size_t)b * N;

  if (tid == 0) { idx[(size_t)b * m] = 0; cur[0] = P[0]; cur[1] = P[1]; cur[2] = P[2]; }
  __syncthreads();
  for (int j = tid; j < N; j += 256) {
    const float dx = P[j*3] - cur[0], dy = P[j*3+1] - cur[1], dz = P[j*3+2] - cur[2];
    d[j] = dx*dx + dy*dy + dz*dz;
  }
  __syncthreads();
  for (int s = 1; s < m; ++s) {
    float best = -1.f; int bidx = 0x7fffffff;
    for (int j = tid; j < N; j += 256) {
      const float v = d[j];
      if (v > best) { best = v; bidx = j; }
    }
    sred[tid] = best; sidx[tid] = bidx;
    __syncthreads();
    for (int off = 128; off > 0; off >>= 1) {
      if (tid < off) {
        if (sred[tid+off] > sred[tid] ||
            (sred[tid+off] == sred[tid] && sidx[tid+off] < sidx[tid])) {
          sred[tid] = sred[tid+off]; sidx[tid] = sidx[tid+off];
        }
      }
      __syncthreads();
    }
    if (tid == 0) {
      const int nx = sidx[0];
      idx[(size_t)b * m + s] = nx;
      cur[0] = P[nx*3]; cur[1] = P[nx*3+1]; cur[2] = P[nx*3+2];
    }
    __syncthreads();
    for (int j = tid; j < N; j += 256) {
      const float dx = P[j*3] - cur[0], dy = P[j*3+1] - cur[1], dz = P[j*3+2] - cur[2];
      d[j] = fminf(d[j], dx*dx + dy*dy + dz*dz);
    }
    __syncthreads();
  }
}

// =========================================================== BatchNorm ======
__global__ void zero_f32(float* p, int n)
{ int i = blockIdx.x * blockDim.x + threadIdx.x; if (i < n) p[i] = 0.f; }

__global__ __launch_bounds__(256)
void bn_stats(const float* __restrict__ x, float* __restrict__ acc, int R, int C)
{
  float s[16], s2[16];
#pragma unroll
  for (int j = 0; j < 16; ++j) { s[j] = 0.f; s2[j] = 0.f; }
  const int lane = threadIdx.x & 31, wav = threadIdx.x >> 5;
  const int nch = C >> 5;
  for (size_t r = (size_t)blockIdx.x * 8 + wav; r < (size_t)R; r += (size_t)gridDim.x * 8) {
    const float* row = x + r * C;
    for (int j = 0; j < nch; ++j) {
      const float v = row[lane + 32 * j];
      s[j] += v; s2[j] += v * v;
    }
  }
  for (int j = 0; j < nch; ++j) {
    atomicAdd(&acc[lane + 32 * j], s[j]);
    atomicAdd(&acc[C + lane + 32 * j], s2[j]);
  }
}

__global__ void bn_apply_relu(const float* __restrict__ x, const float* __restrict__ acc,
                              const float* __restrict__ gamma, const float* __restrict__ beta,
                              float* __restrict__ y, int R, int C)
{
  const size_t i = (size_t)blockIdx.x * blockDim.x + threadIdx.x;
  if (i >= (size_t)R * C) return;
  const int c = (int)(i % C);
  const float mean = acc[c] / R;
  const float var  = acc[C + c] / R - mean * mean;
  const float v = (x[i] - mean) * rsqrtf(var + 1e-5f) * gamma[c] + beta[c];
  y[i] = fmaxf(v, 0.f);
}

// ======================================================= small elementwise ==
__global__ void gather_pos_k(const float* __restrict__ pos, const int* __restrict__ idx,
                             float* __restrict__ out, int N, int m)
{
  const int t = blockIdx.x * blockDim.x + threadIdx.x;     // over B*m
  if (t >= BSZ * m) return;
  const int b = t / m;
  const int src = idx[t];
  const float* p = pos + ((size_t)b * N + src) * 3;
  out[(size_t)t*3+0] = p[0]; out[(size_t)t*3+1] = p[1]; out[(size_t)t*3+2] = p[2];
}

__global__ void gather_max_k(const float* __restrict__ h, const int* __restrict__ nbr,
                             float* __restrict__ out, int N, int m, int C)
{
  const size_t t = (size_t)blockIdx.x * blockDim.x + threadIdx.x;  // B*m*C
  if (t >= (size_t)BSZ * m * C) return;
  const int c = (int)(t % C);
  const int i = (int)((t / C) % m);
  const int b = (int)(t / ((size_t)C * m));
  const int* nn = nbr + ((size_t)b * m + i) * KNN;
  float mx = -3.0e38f;
#pragma unroll
  for (int k = 0; k < KNN; ++k)
    mx = fmaxf(mx, h[((size_t)b * N + nn[k]) * C + c]);
  out[t] = mx;
}

// rel written zero-padded to stride 32 so the K=3 GEMM needs no load guards
__global__ void build_rel_pad(const float* __restrict__ pos, const int* __restrict__ nbr,
                              float* __restrict__ rel, int g0, int cnt, int n)
{
  const int e = blockIdx.x * blockDim.x + threadIdx.x;     // chunk-local edge
  if (e >= cnt * KNN) return;
  const int g = g0 + e / KNN;
  const int b = g / n, i = g % n;
  const int j = nbr[(size_t)g * KNN + (e & (KNN - 1))];
  const float* pi = pos + ((size_t)b * n + i) * 3;
  const float* pj = pos + ((size_t)b * n + j) * 3;
  v4f* out = (v4f*)(rel + ((size_t)e << 5));
  v4f first; first[0] = pi[0]-pj[0]; first[1] = pi[1]-pj[1]; first[2] = pi[2]-pj[2]; first[3] = 0.f;
  out[0] = first;
  const v4f z = {0.f, 0.f, 0.f, 0.f};
#pragma unroll
  for (int q = 1; q < 8; ++q) out[q] = z;
}

__global__ void build_attn_in(const float* __restrict__ adst, const float* __restrict__ asrc,
                              const int* __restrict__ nbr, const float* __restrict__ delta,
                              float* __restrict__ out, int g0, int cnt, int n, int C)
{
  const size_t t = (size_t)blockIdx.x * blockDim.x + threadIdx.x;  // cnt*KNN*C
  if (t >= (size_t)cnt * KNN * C) return;
  const int c = (int)(t % C);
  const size_t e = t / C;
  const int g = g0 + (int)(e / KNN);
  const int b = g / n;
  const int j = nbr[(size_t)g * KNN + (int)(e & (KNN - 1))];
  out[t] = adst[(size_t)g * C + c] - asrc[((size_t)b * n + j) * C + c] + delta[t];
}

// softmax over k per (node,channel), then sum attn*(v_j + delta)
__global__ void aggregate_k(const float* __restrict__ alpha, const float* __restrict__ delta,
                            const float* __restrict__ v, const int* __restrict__ nbr,
                            float* __restrict__ out, int g0, int cnt, int n, int C)
{
  const size_t t = (size_t)blockIdx.x * blockDim.x + threadIdx.x;  // cnt*C
  if (t >= (size_t)cnt * C) return;
  const int c = (int)(t % C);
  const int gl = (int)(t / C);
  const int g = g0 + gl;
  const int b = g / n;
  const int* nn = nbr + (size_t)g * KNN;
  float w[KNN], mx = -3.0e38f;
#pragma unroll
  for (int k = 0; k < KNN; ++k) {
    w[k] = alpha[((size_t)gl * KNN + k) * C + c];
    mx = fmaxf(mx, w[k]);
  }
  float se = 0.f;
#pragma unroll
  for (int k = 0; k < KNN; ++k) { w[k] = __expf(w[k] - mx); se += w[k]; }
  const float inv = 1.f / se;
  float acc = 0.f;
#pragma unroll
  for (int k = 0; k < KNN; ++k) {
    const size_t e = (size_t)gl * KNN + k;
    acc += w[k] * inv * (v[((size_t)b * n + nn[k]) * C + c] + delta[e * C + c]);
  }
  out[(size_t)g * C + c] = acc;
}

__global__ void mean_pool_k(const float* __restrict__ h, float* __restrict__ g, int n, int C)
{
  const int t = blockIdx.x * blockDim.x + threadIdx.x;     // B*C
  if (t >= BSZ * C) return;
  const int b = t / C, c = t % C;
  float s = 0.f;
  for (int i = 0; i < n; ++i) s += h[((size_t)b * n + i) * C + c];
  g[t] = s / n;
}

// ================================================================ host ======
static inline int cdiv(int a, int b) { return (a + b - 1) / b; }
static inline unsigned cdivz(size_t a, int b) { return (unsigned)((a + (size_t)b - 1) / b); }

struct PW { const _Float16* wf; int Npad; };

extern "C" void kernel_launch(void* const* d_in, const int* in_sizes, int n_in,
                              void* d_out, int out_size, void* d_ws, size_t ws_size,
                              hipStream_t stream)
{
  (void)in_sizes; (void)ws_size; (void)out_size;
  if (n_in < 101) return;
  auto P = [&](int i) { return (const float*)d_in[i]; };

  const int Ns[5] = {4096, 1024, 256, 64, 16};
  const int Cs[5] = {32, 64, 128, 256, 512};

  // ---- workspace bump allocator (deterministic layout each call) ----
  char* base = (char*)d_ws;
  size_t off = 0;
  auto allocb = [&](size_t bytes) { char* p = base + off;
                                    off += (bytes + 255) & ~(size_t)255; return p; };
  auto allocf = [&](size_t n) { return (float*)allocb(n * 4); };
  auto alloci = [&](size_t n) { return (int*)allocb(n * 4); };
  auto alloch = [&](size_t n) { return (_Float16*)allocb(n * 2); };

  // ---- pack all weights to WMMA fragment order (f16, padded) ----
  auto packW = [&](int idx, int K, int N) -> PW {
    const int Ks = ((K + 31) / 32) * 32;
    const int Npad = ((N + 15) / 16) * 16;
    const size_t total = (size_t)(Ks / 32) * Npad * 32;
    _Float16* wf = alloch(total);
    pack_w_f16<<<cdivz(total, 256), 256, 0, stream>>>(P(idx), wf, K, N, Npad, (int)total);
    return PW{wf, Npad};
  };

  struct BlockPk {
    PW lin_in, lin, lin_src, lin_dst, lin_out, pos0, pos1, attn0, attn1;
    const float *lin_in_b, *lin_out_b, *pos0_b, *pos1_b, *attn0_b, *attn1_b;
  };
  auto packBlock = [&](int b, int c) {
    BlockPk q;
    q.attn0_b = P(b+0);  q.attn0 = packW(b+1, c, 64);
    q.attn1_b = P(b+2);  q.attn1 = packW(b+3, 64, c);
    q.lin     = packW(b+4, c, c);
    q.lin_dst = packW(b+5, c, c);
    q.lin_in_b = P(b+6);  q.lin_in = packW(b+7, c, c);
    q.lin_out_b = P(b+8); q.lin_out = packW(b+9, c, c);
    q.lin_src = packW(b+10, c, c);
    q.pos0_b = P(b+11); q.pos0 = packW(b+12, 3, 64);
    q.pos1_b = P(b+13); q.pos1 = packW(b+14, 64, c);
    return q;
  };

  PW pw_down[4];
  for (int i = 0; i < 4; ++i) pw_down[i] = packW(2 + 4 * i + 3, Cs[i], Cs[i + 1]);
  PW pw_in = packW(21, 3, 32);
  PW pw_o0 = packW(23, 512, 64);
  PW pw_o1 = packW(25, 64, 10);
  BlockPk blk[5];
  blk[0] = packBlock(86, 32);                       // tr_input
  for (int i = 0; i < 4; ++i) blk[i + 1] = packBlock(26 + 15 * i, Cs[i + 1]);

  // ---- activations / graph buffers ----
  float* posbuf[5]; posbuf[0] = nullptr;
  for (int i = 1; i < 5; ++i) posbuf[i] = allocf((size_t)BSZ * Ns[i] * 3);
  float* hbuf[5];
  for (int i = 0; i < 5; ++i) hbuf[i] = allocf((size_t)BSZ * Ns[i] * Cs[i]);
  int* nbrbuf[5];
  for (int i = 0; i < 5; ++i) nbrbuf[i] = alloci((size_t)BSZ * Ns[i] * KNN);
  int* td_nbr = alloci((size_t)BSZ * 1024 * KNN);
  int* fps_idx = alloci((size_t)BSZ * 1024);
  float* fps_d = allocf((size_t)BSZ * 4096);
  float* bnacc = allocf(1024);
  float* xpad = allocf((size_t)BSZ * 4096 * 32);
  const size_t NODE_MAX = (size_t)BSZ * 4096 * 32;
  const size_t TD_MAX   = (size_t)BSZ * 4096 * 64;
  float* nb_lin  = allocf(NODE_MAX);
  float* nb_v    = allocf(NODE_MAX);
  float* nb_asrc = allocf(NODE_MAX);
  float* nb_adst = allocf(NODE_MAX);
  float* nb_agg  = allocf(NODE_MAX);
  float* nb_tmp  = allocf(TD_MAX);
  float* nb_tmp2 = allocf(TD_MAX);
  const size_t EMAX  = (size_t)NRCHUNK * KNN;
  const size_t ECMAX = 8388608;
  float* e_rel   = allocf(EMAX * 32);               // stride-32 zero-padded
  float* e_h64   = allocf(EMAX * 64);
  float* e_delta = allocf(ECMAX);
  float* e_attn  = allocf(ECMAX);
  float* e_alpha = allocf(ECMAX);
  float* gpool = allocf((size_t)BSZ * 512);
  float* ghid  = allocf((size_t)BSZ * 64);

  auto gemm = [&](const float* A, int Ks, PW pw, const float* b, float* C,
                  int M, int Nd, int relu) {
    dim3 grid(cdiv(cdiv(M, 16), 4), cdiv(pw.Npad, 32));
    gemm_wmma_fast<<<grid, dim3(128), 0, stream>>>(A, pw.wf, b, C, M, Ks, Nd, pw.Npad, relu);
  };
  auto run_bn = [&](const float* x, const float* gamma, const float* beta,
                    float* y, int R, int C) {
    zero_f32<<<cdiv(2 * C, 256), 256, 0, stream>>>(bnacc, 2 * C);
    bn_stats<<<64, 256, 0, stream>>>(x, bnacc, R, C);
    bn_apply_relu<<<cdivz((size_t)R * C, 256), 256, 0, stream>>>(x, bnacc, gamma, beta, y, R, C);
  };
  auto run_block = [&](const BlockPk& q, float* h, const float* pos,
                       const int* nbr, int n, int c) {
    const int R = BSZ * n;
    gemm(h,      c, q.lin_in,  q.lin_in_b, nb_lin,  R, c, 1);  // x = relu(lin_in h)
    gemm(nb_lin, c, q.lin,     nullptr,    nb_v,    R, c, 0);
    gemm(nb_lin, c, q.lin_src, nullptr,    nb_asrc, R, c, 0);
    gemm(nb_lin, c, q.lin_dst, nullptr,    nb_adst, R, c, 0);
    for (int g0 = 0; g0 < R; g0 += NRCHUNK) {                  // L2-resident chunk
      const int cnt = (R - g0 < NRCHUNK) ? (R - g0) : NRCHUNK;
      const int E = cnt * KNN;
      build_rel_pad<<<cdiv(E, 256), 256, 0, stream>>>(pos, nbr, e_rel, g0, cnt, n);
      gemm(e_rel, 32, q.pos0, q.pos0_b, e_h64, E, 64, 1);
      gemm(e_h64, 64, q.pos1, q.pos1_b, e_delta, E, c, 1);     // delta
      build_attn_in<<<cdivz((size_t)E * c, 256), 256, 0, stream>>>(
          nb_adst, nb_asrc, nbr, e_delta, e_attn, g0, cnt, n, c);
      gemm(e_attn, c, q.attn0, q.attn0_b, e_h64, E, 64, 1);
      gemm(e_h64, 64, q.attn1, q.attn1_b, e_alpha, E, c, 1);   // alpha
      aggregate_k<<<cdivz((size_t)cnt * c, 256), 256, 0, stream>>>(
          e_alpha, e_delta, nb_v, nbr, nb_agg, g0, cnt, n, c);
    }
    gemm(nb_agg, c, q.lin_out, q.lin_out_b, h, R, c, 1);       // relu(lin_out)
  };

  const float* x    = P(0);
  const float* pos0 = P(1);

  // ---------------- input MLP + BN + ReLU ----------------
  pad_rows3<<<cdiv(BSZ * 4096 * 32, 256), 256, 0, stream>>>(x, xpad, BSZ * 4096);
  gemm(xpad, 32, pw_in, P(20), nb_tmp, BSZ * 4096, 32, 0);
  run_bn(nb_tmp, P(19), P(18), hbuf[0], BSZ * 4096, 32);       // gamma=19, beta=18

  // ---------------- tr_input ----------------
  knn_kernel<<<dim3(cdiv(4096, 256), BSZ), 256, 0, stream>>>(pos0, pos0, nbrbuf[0], 4096, 4096, 1);
  run_block(blk[0], hbuf[0], pos0, nbrbuf[0], 4096, 32);

  // ---------------- 4 x (transition_down + tr_down) ----------------
  const float* posc = pos0;
  for (int i = 0; i < 4; ++i) {
    const int n = Ns[i], m = Ns[i + 1], co = Cs[i + 1];
    const int db = 2 + 4 * i;                                  // beta,gamma,lin.b,lin.w
    gemm(hbuf[i], Cs[i], pw_down[i], P(db + 2), nb_tmp, BSZ * n, co, 0);
    run_bn(nb_tmp, P(db + 1), P(db + 0), nb_tmp2, BSZ * n, co);
    fps_kernel<<<BSZ, 256, 0, stream>>>(posc, fps_idx, fps_d, n, m);
    gather_pos_k<<<cdiv(BSZ * m, 256), 256, 0, stream>>>(posc, fps_idx, posbuf[i + 1], n, m);
    knn_kernel<<<dim3(cdiv(m, 256), BSZ), 256, 0, stream>>>(posbuf[i + 1], posc, td_nbr, m, n, 0);
    gather_max_k<<<cdivz((size_t)BSZ * m * co, 256), 256, 0, stream>>>(
        nb_tmp2, td_nbr, hbuf[i + 1], n, m, co);
    knn_kernel<<<dim3(cdiv(m, 256), BSZ), 256, 0, stream>>>(posbuf[i + 1], posbuf[i + 1],
                                                            nbrbuf[i + 1], m, m, 1);
    run_block(blk[i + 1], hbuf[i + 1], posbuf[i + 1], nbrbuf[i + 1], m, co);
    posc = posbuf[i + 1];
  }

  // ---------------- pool + classifier ----------------
  mean_pool_k<<<cdiv(BSZ * 512, 256), 256, 0, stream>>>(hbuf[4], gpool, 16, 512);
  gemm(gpool, 512, pw_o0, P(22), ghid, BSZ, 64, 1);            // mlp_out[0] + ReLU
  gemm(ghid, 64, pw_o1, P(24), (float*)d_out, BSZ, 10, 0);     // mlp_out[1]
}
